// Transformer_89790586290822
// MI455X (gfx1250) — matrix operand order repaired and checked by reading, not compiled, not run on hardware
//
#include <hip/hip_runtime.h>
#include <hip/hip_bf16.h>

// ---------------------------------------------------------------------------
// Types for CDNA5 WMMA (gfx1250, wave32)
// ---------------------------------------------------------------------------
typedef __bf16 v16bf __attribute__((ext_vector_type(16)));
typedef float  v8f   __attribute__((ext_vector_type(8)));

struct alignas(16) U128 { unsigned int w[4]; };

struct Frag {
    union { U128 u[2]; v16bf v; };
};

__device__ __forceinline__ unsigned short f2bf(float f) {
    union { float f; unsigned u; } c; c.f = f;
    unsigned r = c.u + 0x7FFFu + ((c.u >> 16) & 1u);   // round-to-nearest-even
    return (unsigned short)(r >> 16);
}
__device__ __forceinline__ float bf2f(unsigned short u) {
    union { unsigned u; float f; } c; c.u = ((unsigned)u) << 16;
    return c.f;
}
// Low 32 bits of a generic pointer to LDS == LDS byte address (ISA: LDS aperture
// maps addr[31:0] to LDS).
__device__ __forceinline__ unsigned lds_addr(const void* p) {
    return (unsigned)(unsigned long long)p;
}
// Async DMA: global -> LDS, 16 bytes per lane, tracked by ASYNCcnt.
__device__ __forceinline__ void async_b128(unsigned lds, unsigned goff,
                                           const void* base) {
    asm volatile("global_load_async_to_lds_b128 %0, %1, %2"
                 :: "v"(lds), "v"(goff), "s"(base) : "memory");
}

// A-matrix fragment (16x32 bf16), ISA 7.12.2 layout:
// lane: m = lane&15, h = lane>>4; VGPR0-3 = K[8h..8h+7], VGPR4-7 = K[16+8h..+7]
__device__ __forceinline__ void load_a(Frag& f, const unsigned short* base,
                                       int stride, int lane) {
    int m = lane & 15, h = lane >> 4;
    const unsigned short* p = base + m * stride + 8 * h;
    f.u[0] = *(const U128*)(p);
    f.u[1] = *(const U128*)(p + 16);
}
// B-matrix fragment (32x16 bf16), LDS stored K-major per column n:
// lane: n = lane&15, h = lane>>4; VGPR0-3 = K[8h..8h+7], VGPR4-7 = K[16+8h..+7]
__device__ __forceinline__ void load_b(Frag& f, const unsigned short* base,
                                       int stride, int lane) {
    int n = lane & 15, h = lane >> 4;
    const unsigned short* p = base + n * stride + 8 * h;
    f.u[0] = *(const U128*)(p);
    f.u[1] = *(const U128*)(p + 16);
}
__device__ __forceinline__ v8f wmma_bf16(const Frag& a, const Frag& b, v8f c) {
    return __builtin_amdgcn_wmma_f32_16x16x32_bf16(
        false, a.v, false, b.v, (short)0, c, false, false);
}

// ---------------------------------------------------------------------------
// Weight convert + transpose: W[K][N] f32  ->  Wt[N][K] bf16  (K-major B operand)
// 32x32 tiles, 256 threads, coalesced both sides.
// ---------------------------------------------------------------------------
__global__ __launch_bounds__(256) void w_transpose(const float* __restrict__ W,
                                                   unsigned short* __restrict__ Wt,
                                                   int K, int N) {
    __shared__ float t[32][33];
    int bx = blockIdx.x * 32, by = blockIdx.y * 32;     // bx: n, by: k
    int tx = threadIdx.x & 31, ty = threadIdx.x >> 5;   // 32 x 8
    for (int i = 0; i < 4; ++i)
        t[ty + 8 * i][tx] = W[(size_t)(by + ty + 8 * i) * N + bx + tx];
    __syncthreads();
    for (int i = 0; i < 4; ++i)
        Wt[(size_t)(bx + ty + 8 * i) * K + by + tx] = f2bf(t[tx][ty + 8 * i]);
}

// ---------------------------------------------------------------------------
// Embedding gather + sinusoidal positional encoding (bf16 output)
// ---------------------------------------------------------------------------
__global__ __launch_bounds__(128) void embed_pe(const int* __restrict__ tok,
                                                const float* __restrict__ emb,
                                                unsigned short* __restrict__ out,
                                                int seqlen) {
    int idx = blockIdx.x;                   // b*seqlen + s
    int s   = idx % seqlen;
    int t   = tok[idx];
    const float k = 0.01799331553f;         // ln(10000)/512
    for (int d = threadIdx.x; d < 512; d += 128) {
        float freq = __expf(-(float)(d & ~1) * k);
        float ang  = (float)s * freq;
        float pe   = (d & 1) ? __cosf(ang) : __sinf(ang);
        out[(size_t)idx * 512 + d] =
            f2bf(emb[(size_t)t * 512 + d] * 22.62741699797f + pe);
    }
}

// ---------------------------------------------------------------------------
// Tiled bf16 WMMA GEMM with async-DMA double buffering:
//   C[M,N] = act(A[M,K](bf16) @ Wt[N,K](bf16)^T + bias[N])
// Block: 256 threads = 8 waves; tile 128x128, K-step 32, 2-stage pipeline.
// ---------------------------------------------------------------------------
#define LSTR 40   // LDS row stride (elems): 80 B, 16-B aligned, conflict-padded

__global__ __launch_bounds__(256) void gemm_bf16(const unsigned short* __restrict__ A,
                                                 const unsigned short* __restrict__ Wt,
                                                 const float* __restrict__ bias,
                                                 void* __restrict__ C,
                                                 int M, int N, int K,
                                                 int relu, int out_f32) {
    __shared__ unsigned short Alds[2][128 * LSTR];   // [m][k]
    __shared__ unsigned short Blds[2][128 * LSTR];   // [n][k]

    int tid  = threadIdx.x;
    int lane = tid & 31, wave = tid >> 5;
    int wm   = wave >> 1, wn = wave & 1;
    int mbase = blockIdx.y * 128, nbase = blockIdx.x * 128;

    unsigned albase[2] = { lds_addr(&Alds[0][0]), lds_addr(&Alds[1][0]) };
    unsigned blbase[2] = { lds_addr(&Blds[0][0]), lds_addr(&Blds[1][0]) };

    v8f acc[2][4] = {};

    // per-thread DMA chunk coords: tile = 128 rows x 32 k = 512 x 16B chunks
    int c0row = tid >> 1, c0ch = (tid & 1);          // chunks 0..255 -> rows 0..127, ch 0..1
    // second chunk: +2 in ch (same row)
    auto issue_tile = [&](int kt, int buf) {
        for (int i = 0; i < 2; ++i) {
            int ch = c0ch + 2 * i;                   // 0..3 (8 elems each)
            unsigned lo = albase[buf] + c0row * (LSTR * 2) + ch * 16;
            unsigned go = (unsigned)(((size_t)(mbase + c0row) * K + kt + ch * 8) * 2);
            async_b128(lo, go, A);
        }
        for (int i = 0; i < 2; ++i) {
            int ch = c0ch + 2 * i;
            unsigned lo = blbase[buf] + c0row * (LSTR * 2) + ch * 16;
            unsigned go = (unsigned)(((size_t)(nbase + c0row) * K + kt + ch * 8) * 2);
            async_b128(lo, go, Wt);
        }
    };

    int nk = K / 32;
    issue_tile(0, 0);

    for (int t = 0; t < nk; ++t) {
        __syncthreads();                              // prev compute done everywhere
        if (t + 1 < nk) {
            issue_tile((t + 1) * 32, (t + 1) & 1);
            asm volatile("s_wait_asynccnt 0x4" ::: "memory");  // tile t landed
        } else {
            asm volatile("s_wait_asynccnt 0x0" ::: "memory");
        }
        __syncthreads();                              // tile t visible to all waves

        const unsigned short* al = &Alds[t & 1][0];
        const unsigned short* bl = &Blds[t & 1][0];
        Frag a0, a1, b[4];
        load_a(a0, al + (wm * 32 + 0)  * LSTR, LSTR, lane);
        load_a(a1, al + (wm * 32 + 16) * LSTR, LSTR, lane);
        for (int j = 0; j < 4; ++j)
            load_b(b[j], bl + (wn * 64 + j * 16) * LSTR, LSTR, lane);
        for (int j = 0; j < 4; ++j) {
            acc[0][j] = wmma_bf16(a0, b[j], acc[0][j]);
            acc[1][j] = wmma_bf16(a1, b[j], acc[1][j]);
        }
    }

    int h = lane >> 4, nl = lane & 15;
    for (int im = 0; im < 2; ++im)
        for (int jn = 0; jn < 4; ++jn) {
            int ncol  = nbase + wn * 64 + jn * 16 + nl;
            float bv  = bias[ncol];
            int mrow0 = mbase + wm * 32 + im * 16 + 8 * h;
            for (int r = 0; r < 8; ++r) {
                float v = acc[im][jn][r] + bv;
                if (relu) v = fmaxf(v, 0.f);
                size_t idx = (size_t)(mrow0 + r) * N + ncol;
                if (out_f32) ((float*)C)[idx] = v;
                else ((unsigned short*)C)[idx] = f2bf(v);
            }
        }
}

// ---------------------------------------------------------------------------
// Flash attention: scores never leave the WGP. All operands bf16 in global.
// Q/K/V/O: [B, len, H, 64]. Block: 128 thr = 4 waves; 64 q/block; 32-key tiles.
// K tile is staged by async DMA; V needs a 2-byte transpose (manual); Q is
// scaled by 1/sqrt(dk) during staging.
// ---------------------------------------------------------------------------
__global__ __launch_bounds__(128) void flash_attn(const unsigned short* __restrict__ Q,
                                                  const unsigned short* __restrict__ Kin,
                                                  const unsigned short* __restrict__ Vin,
                                                  unsigned short* __restrict__ O,
                                                  int qlen, int kvlen, int causal) {
    __shared__ unsigned short Qlds[64 * 72];       // [q][dk]   stride 72 (144 B)
    __shared__ unsigned short Klds[32 * 72];       // [key][dk]
    __shared__ unsigned short Vlds[64 * LSTR];     // [dk][key] (K-major B operand)
    __shared__ unsigned short Plds[4 * 16 * LSTR]; // per-wave P tile 16x32

    int tid = threadIdx.x, lane = tid & 31, wave = tid >> 5;
    int b = blockIdx.z, h = blockIdx.y;
    int qtile = blockIdx.x * 64;
    const unsigned short* Qp = Q   + (size_t)b * qlen  * 512 + h * 64;
    const unsigned short* Kp = Kin + (size_t)b * kvlen * 512 + h * 64;
    const unsigned short* Vp = Vin + (size_t)b * kvlen * 512 + h * 64;
    unsigned kldsb = lds_addr(&Klds[0]);

    // stage Q (scale by 0.125), 64x64 bf16, 16B chunks
    for (int i = 0; i < 4; ++i) {
        int c = tid + 128 * i;                       // 0..511
        int row = c >> 3, ch = c & 7;
        U128 q = *(const U128*)(Qp + (size_t)(qtile + row) * 512 + ch * 8);
        U128 o;
        for (int j = 0; j < 4; ++j) {
            unsigned u = q.w[j];
            float lo = bf2f((unsigned short)u) * 0.125f;
            float hi = bf2f((unsigned short)(u >> 16)) * 0.125f;
            o.w[j] = (unsigned)f2bf(lo) | ((unsigned)f2bf(hi) << 16);
        }
        *(U128*)(Qlds + row * 72 + ch * 8) = o;
    }
    __syncthreads();

    Frag aq0, aq1;
    load_a(aq0, Qlds + (wave * 16) * 72,      72, lane);   // dk 0..31
    load_a(aq1, Qlds + (wave * 16) * 72 + 32, 72, lane);   // dk 32..63

    float m_s[8], l_s[8];
    for (int r = 0; r < 8; ++r) { m_s[r] = -3e38f; l_s[r] = 0.f; }
    v8f o[4] = {};

    int hh = lane >> 4, nl = lane & 15;
    int nkt = causal ? (qtile + 64) / 32 : kvlen / 32;

    for (int kt = 0; kt < nkt; ++kt) {
        __syncthreads();                              // tiles free to overwrite
        // K tile 32x64: pure DMA (row copy, 2 chunks/thread)
        for (int i = 0; i < 2; ++i) {
            int c = tid + 128 * i;                    // 0..255
            int row = c >> 3, ch = c & 7;
            unsigned lo = kldsb + row * 144 + ch * 16;
            unsigned go = (unsigned)(((size_t)(kt * 32 + row) * 512 + ch * 8) * 2);
            async_b128(lo, go, Kp);
        }
        // V tile transposed -> Vlds[dk][key] (2-byte scatter; manual)
        {
            int c2 = tid & 31, kr = tid >> 5;
            for (int i = 0; i < 8; ++i) {
                int key = kr + 4 * i;
                unsigned vv = *(const unsigned*)(Vp + (size_t)(kt * 32 + key) * 512 + c2 * 2);
                Vlds[(c2 * 2 + 0) * LSTR + key] = (unsigned short)vv;
                Vlds[(c2 * 2 + 1) * LSTR + key] = (unsigned short)(vv >> 16);
            }
        }
        asm volatile("s_wait_asynccnt 0x0" ::: "memory");
        __syncthreads();

        // S = Q K^T  (two 16-key tiles, two 32-deep k-steps each)
        v8f s[2] = {};
        for (int ns = 0; ns < 2; ++ns) {
            Frag bk;
            load_b(bk, Klds + (ns * 16) * 72,      72, lane);
            s[ns] = wmma_bf16(aq0, bk, s[ns]);
            load_b(bk, Klds + (ns * 16) * 72 + 32, 72, lane);
            s[ns] = wmma_bf16(aq1, bk, s[ns]);
        }
        if (causal) {
            for (int ns = 0; ns < 2; ++ns) {
                int key = kt * 32 + ns * 16 + nl;
                for (int r = 0; r < 8; ++r) {
                    int qg = qtile + wave * 16 + 8 * hh + r;
                    if (key > qg) s[ns][r] = -3e38f;
                }
            }
        }
        // online softmax (row lives across a 16-lane group; shuffle-reduce)
        unsigned short* pl = Plds + wave * 16 * LSTR;
        for (int r = 0; r < 8; ++r) {
            float t = fmaxf(s[0][r], s[1][r]);
            for (int m = 1; m < 16; m <<= 1) t = fmaxf(t, __shfl_xor(t, m, 32));
            float nm = fmaxf(m_s[r], t);
            float sc = __expf(m_s[r] - nm);
            float p0 = __expf(s[0][r] - nm);
            float p1 = __expf(s[1][r] - nm);
            float rs = p0 + p1;
            for (int m = 1; m < 16; m <<= 1) rs += __shfl_xor(rs, m, 32);
            l_s[r] = l_s[r] * sc + rs;
            m_s[r] = nm;
            for (int t4 = 0; t4 < 4; ++t4) o[t4][r] *= sc;
            pl[(8 * hh + r) * LSTR + nl]      = f2bf(p0);   // wave-private; DS in-order
            pl[(8 * hh + r) * LSTR + 16 + nl] = f2bf(p1);
        }
        // O += P V
        Frag ap;
        load_a(ap, pl, LSTR, lane);
        for (int t4 = 0; t4 < 4; ++t4) {
            Frag bv;
            load_b(bv, Vlds + (t4 * 16) * LSTR, LSTR, lane);
            o[t4] = wmma_bf16(ap, bv, o[t4]);
        }
    }

    for (int t4 = 0; t4 < 4; ++t4)
        for (int r = 0; r < 8; ++r) {
            int qg = qtile + wave * 16 + 8 * hh + r;
            O[((size_t)(b * qlen + qg)) * 512 + h * 64 + t4 * 16 + nl] =
                f2bf(o[t4][r] / l_s[r]);
        }
}

// ---------------------------------------------------------------------------
// Fused residual-add + LayerNorm (bf16 in/out, f32 math), one wave per row.
// ---------------------------------------------------------------------------
__global__ __launch_bounds__(256) void add_ln(unsigned short* __restrict__ x,
                                              const unsigned short* __restrict__ br,
                                              const float* __restrict__ sc,
                                              const float* __restrict__ bi) {
    int lane = threadIdx.x & 31, wave = threadIdx.x >> 5;
    size_t row = (size_t)blockIdx.x * 8 + wave;
    float v[16], s = 0.f;
    for (int i = 0; i < 16; ++i) {
        int d = lane + 32 * i;
        v[i] = bf2f(x[row * 512 + d]) + bf2f(br[row * 512 + d]);
        s += v[i];
    }
    for (int m = 1; m < 32; m <<= 1) s += __shfl_xor(s, m, 32);
    float mean = s * (1.f / 512.f);
    float vs = 0.f;
    for (int i = 0; i < 16; ++i) { float d = v[i] - mean; vs += d * d; }
    for (int m = 1; m < 32; m <<= 1) vs += __shfl_xor(vs, m, 32);
    float inv = rsqrtf(vs * (1.f / 512.f) + 1e-5f);
    for (int i = 0; i < 16; ++i) {
        int d = lane + 32 * i;
        x[row * 512 + d] = f2bf((v[i] - mean) * inv * sc[d] + bi[d]);
    }
}

// ---------------------------------------------------------------------------
// Host orchestration
// ---------------------------------------------------------------------------
extern "C" void kernel_launch(void* const* d_in, const int* in_sizes, int n_in,
                              void* d_out, int out_size, void* d_ws, size_t ws_size,
                              hipStream_t stream) {
    (void)in_sizes; (void)n_in; (void)out_size; (void)ws_size;
    const int B = 4, S = 1024, T = 1024, D = 512, F = 2048, L = 6, H = 8, VT = 32000;

    const int*   src     = (const int*)d_in[0];
    const int*   tgt     = (const int*)d_in[1];
    const float* src_emb = (const float*)d_in[2];
    const float* tgt_emb = (const float*)d_in[3];
    const float* eaw     = (const float*)d_in[4];
    const float* eab     = (const float*)d_in[5];
    const float* ew1     = (const float*)d_in[6];
    const float* eb1     = (const float*)d_in[7];
    const float* ew2     = (const float*)d_in[8];
    const float* eb2     = (const float*)d_in[9];
    const float* elns    = (const float*)d_in[10];
    const float* elnb    = (const float*)d_in[11];
    const float* saw     = (const float*)d_in[12];
    const float* sab     = (const float*)d_in[13];
    const float* caw     = (const float*)d_in[14];
    const float* cab     = (const float*)d_in[15];
    const float* dw1     = (const float*)d_in[16];
    const float* db1     = (const float*)d_in[17];
    const float* dw2     = (const float*)d_in[18];
    const float* db2     = (const float*)d_in[19];
    const float* dlns    = (const float*)d_in[20];
    const float* dlnb    = (const float*)d_in[21];
    const float* out_w   = (const float*)d_in[22];
    const float* out_b   = (const float*)d_in[23];

    unsigned short* ws = (unsigned short*)d_ws;
    const size_t NA = (size_t)B * S * D;            // 2M elems / activation buffer
    unsigned short* xb   = ws;
    unsigned short* yb   = ws + 1 * NA;
    unsigned short* qb   = ws + 2 * NA;
    unsigned short* kb   = ws + 3 * NA;
    unsigned short* vb   = ws + 4 * NA;
    unsigned short* ctxb = ws + 5 * NA;
    unsigned short* pb   = ws + 6 * NA;
    unsigned short* fb   = ws + 7 * NA;             // B*S*F
    unsigned short* walloc = ws + 7 * NA + (size_t)B * S * F;

    const size_t DDsz = (size_t)D * D, DFsz = (size_t)D * F;
    auto walk = [&](size_t n) { unsigned short* p = walloc; walloc += n; return p; };

    unsigned short *tEA[6][4], *tE1[6], *tE2[6];
    unsigned short *tSA[6][4], *tCA[6][4], *tD1[6], *tD2[6], *tOW;

    auto xpose = [&](const float* W, unsigned short* Wt, int K, int N) {
        w_transpose<<<dim3(N / 32, K / 32), 256, 0, stream>>>(W, Wt, K, N);
    };

    for (int l = 0; l < L; ++l) {
        for (int p = 0; p < 4; ++p) {
            tEA[l][p] = walk(DDsz); xpose(eaw + ((size_t)l * 4 + p) * DDsz, tEA[l][p], D, D);
            tSA[l][p] = walk(DDsz); xpose(saw + ((size_t)l * 4 + p) * DDsz, tSA[l][p], D, D);
            tCA[l][p] = walk(DDsz); xpose(caw + ((size_t)l * 4 + p) * DDsz, tCA[l][p], D, D);
        }
        tE1[l] = walk(DFsz); xpose(ew1 + (size_t)l * DFsz, tE1[l], D, F);
        tE2[l] = walk(DFsz); xpose(ew2 + (size_t)l * DFsz, tE2[l], F, D);
        tD1[l] = walk(DFsz); xpose(dw1 + (size_t)l * DFsz, tD1[l], D, F);
        tD2[l] = walk(DFsz); xpose(dw2 + (size_t)l * DFsz, tD2[l], F, D);
    }
    tOW = walk((size_t)D * VT); xpose(out_w, tOW, D, VT);

    const int M = B * S;
    auto gemm = [&](const unsigned short* A, const unsigned short* Wt,
                    const float* bias, void* C, int Nn, int Kk, int relu, int f32o) {
        gemm_bf16<<<dim3(Nn / 128, M / 128), 256, 0, stream>>>(A, Wt, bias, C,
                                                               M, Nn, Kk, relu, f32o);
    };

    embed_pe<<<B * S, 128, 0, stream>>>(src, src_emb, xb, S);
    embed_pe<<<B * T, 128, 0, stream>>>(tgt, tgt_emb, yb, T);

    // ------------------- encoder -------------------
    for (int l = 0; l < L; ++l) {
        const float* bp = eab + (size_t)l * 4 * D;
        gemm(xb, tEA[l][0], bp + 0 * D, qb, D, D, 0, 0);
        gemm(xb, tEA[l][1], bp + 1 * D, kb, D, D, 0, 0);
        gemm(xb, tEA[l][2], bp + 2 * D, vb, D, D, 0, 0);
        flash_attn<<<dim3(S / 64, H, B), 128, 0, stream>>>(qb, kb, vb, ctxb, S, S, 0);
        gemm(ctxb, tEA[l][3], bp + 3 * D, pb, D, D, 0, 0);
        add_ln<<<M / 8, 256, 0, stream>>>(xb, pb, elns + (size_t)(l * 2 + 0) * D,
                                                  elnb + (size_t)(l * 2 + 0) * D);
        gemm(xb, tE1[l], eb1 + (size_t)l * F, fb, F, D, 1, 0);
        gemm(fb, tE2[l], eb2 + (size_t)l * D, pb, D, F, 0, 0);
        add_ln<<<M / 8, 256, 0, stream>>>(xb, pb, elns + (size_t)(l * 2 + 1) * D,
                                                  elnb + (size_t)(l * 2 + 1) * D);
    }

    // ------------------- decoder -------------------
    for (int l = 0; l < L; ++l) {
        const float* sb  = sab + (size_t)l * 4 * D;
        const float* cb2 = cab + (size_t)l * 4 * D;
        gemm(yb, tSA[l][0], sb + 0 * D, qb, D, D, 0, 0);
        gemm(yb, tSA[l][1], sb + 1 * D, kb, D, D, 0, 0);
        gemm(yb, tSA[l][2], sb + 2 * D, vb, D, D, 0, 0);
        flash_attn<<<dim3(T / 64, H, B), 128, 0, stream>>>(qb, kb, vb, ctxb, T, T, 1);
        gemm(ctxb, tSA[l][3], sb + 3 * D, pb, D, D, 0, 0);
        add_ln<<<M / 8, 256, 0, stream>>>(yb, pb, dlns + (size_t)(l * 3 + 0) * D,
                                                  dlnb + (size_t)(l * 3 + 0) * D);
        gemm(yb, tCA[l][0], cb2 + 0 * D, qb, D, D, 0, 0);
        gemm(xb, tCA[l][1], cb2 + 1 * D, kb, D, D, 0, 0);
        gemm(xb, tCA[l][2], cb2 + 2 * D, vb, D, D, 0, 0);
        flash_attn<<<dim3(T / 64, H, B), 128, 0, stream>>>(qb, kb, vb, ctxb, T, S, 0);
        gemm(ctxb, tCA[l][3], cb2 + 3 * D, pb, D, D, 0, 0);
        add_ln<<<M / 8, 256, 0, stream>>>(yb, pb, dlns + (size_t)(l * 3 + 1) * D,
                                                  dlnb + (size_t)(l * 3 + 1) * D);
        gemm(yb, tD1[l], db1 + (size_t)l * F, fb, F, D, 1, 0);
        gemm(fb, tD2[l], db2 + (size_t)l * D, pb, D, F, 0, 0);
        add_ln<<<M / 8, 256, 0, stream>>>(yb, pb, dlns + (size_t)(l * 3 + 2) * D,
                                                  dlnb + (size_t)(l * 3 + 2) * D);
    }

    // ------------------- output projection (f32 logits) -------------------
    gemm(yb, tOW, out_b, d_out, VT, D, 0, 1);
}
